// VectorQuantizer_32195074851360
// MI455X (gfx1250) — matrix operand order repaired and checked by reading, not compiled, not run on hardware
//
#include <hip/hip_runtime.h>
#include <hip/hip_bf16.h>

// ---------------------------------------------------------------------------
// VQ: distances = ||x||^2 + ||e||^2 - 2 x.e ; argmin over K=1024 ; gather.
// B=32, C=256, H=W=32  ->  N = 32768 rows of depth 256 vs K = 1024 codes.
// v_wmma_f32_16x16x32_bf16 for the 17.2-GFLOP distance matmul; A fragments
// pinned in VGPRs; codebook blocks streamed to LDS with the CDNA5 async
// global->LDS path (ASYNCcnt); gather output uses the exact fp32 codebook.
// ---------------------------------------------------------------------------

typedef __bf16 bf16_t;
typedef __attribute__((ext_vector_type(8)))  __bf16 v8bf;
typedef __attribute__((ext_vector_type(16))) __bf16 v16bf;
typedef __attribute__((ext_vector_type(8)))  float  v8f;

#define NUM_EMB   1024
#define EMB_DIM   256
#define N_ROWS    32768      // 32 * 32 * 32
#define SPATIAL   1024       // 32 * 32
#define ROWS_PER_WG 128
#define PITCH     264        // 256 halves + 8 pad (16B) -> bank-spread rows
#define NBLK      (NUM_EMB / 16)   // 64 column blocks

#ifndef __has_builtin
#define __has_builtin(x) 0
#endif
#if __has_builtin(__builtin_amdgcn_global_load_async_to_lds_b128) && \
    __has_builtin(__builtin_amdgcn_s_wait_asynccnt)
#define HAVE_ASYNC 1
#else
#define HAVE_ASYNC 0
#endif

#if HAVE_ASYNC
// Builtin params are v4i pointers in AS1 (global) / AS3 (LDS), per hipcc
// diagnostic: "__attribute__((__vector_size__(4*sizeof(int)))) int __device__ *"
typedef int v4i_gcc __attribute__((vector_size(16)));
typedef __attribute__((address_space(1))) v4i_gcc GV4;
typedef __attribute__((address_space(3))) v4i_gcc LV4;
__device__ __forceinline__ void async_cp_b128(const void* g, void* l) {
    __builtin_amdgcn_global_load_async_to_lds_b128(
        (GV4*)(void*)g, (LV4*)l, 0, 0);
}
#endif

// ---------------------------------------------------------------------------
// Kernel 0: codebook fp32 -> bf16 into ws, plus exact fp32 column norms.
// ---------------------------------------------------------------------------
__global__ __launch_bounds__(32)
void vq_prep_kernel(const float* __restrict__ weight,
                    bf16_t* __restrict__ wB,
                    float* __restrict__ cn2) {
    const int k = blockIdx.x;
    const int t = threadIdx.x;           // 0..31, each handles 8 floats
    const float4* src = (const float4*)(weight + (size_t)k * EMB_DIM + t * 8);
    float4 f0 = src[0];
    float4 f1 = src[1];
    float s = f0.x * f0.x + f0.y * f0.y + f0.z * f0.z + f0.w * f0.w +
              f1.x * f1.x + f1.y * f1.y + f1.z * f1.z + f1.w * f1.w;
    v8bf h;
    h[0] = (__bf16)f0.x; h[1] = (__bf16)f0.y; h[2] = (__bf16)f0.z; h[3] = (__bf16)f0.w;
    h[4] = (__bf16)f1.x; h[5] = (__bf16)f1.y; h[6] = (__bf16)f1.z; h[7] = (__bf16)f1.w;
    *(v8bf*)(wB + (size_t)k * EMB_DIM + t * 8) = h;
    #pragma unroll
    for (int off = 16; off >= 1; off >>= 1)
        s += __shfl_xor(s, off, 32);
    if (t == 0) cn2[k] = s;
}

// ---------------------------------------------------------------------------
// Kernel 1: main. grid = 256 WGs x 256 threads (8 wave32's).
// ---------------------------------------------------------------------------
__global__ __launch_bounds__(256)
void vq_main_kernel(const float* __restrict__ in,       // [32,256,32,32]
                    const float* __restrict__ weight,   // [1024,256] fp32
                    const bf16_t* __restrict__ wB,      // [1024,256] bf16
                    const float* __restrict__ cn2,      // [1024]
                    float* __restrict__ outQ,           // [32,256,32,32]
                    int* __restrict__ outIdx) {         // [32768]
    __shared__ __align__(16) bf16_t sA[ROWS_PER_WG * PITCH];   // 66 KB
    __shared__ __align__(16) bf16_t sB[2][16 * PITCH];         // 16.5 KB
    __shared__ float sPart[256];
    __shared__ float sRowN[ROWS_PER_WG];
    __shared__ int   sIdx[ROWS_PER_WG];

    const int t    = threadIdx.x;
    const int wave = t >> 5;
    const int lane = t & 31;
    const int half = lane >> 4;      // 0 | 1 (lane group)
    const int m    = lane & 15;      // row (A) / column (B) inside tile

    const int rBase = blockIdx.x * ROWS_PER_WG;   // global row base (same b)
    const int b     = rBase >> 10;
    const int sp0   = rBase & (SPATIAL - 1);
    const float* inB = in + (size_t)b * EMB_DIM * SPATIAL;

    // B-block copy geometry for this thread: row 0..15, 32B slice 0..15
    const int bRowT = t >> 4;
    const int bColT = (t & 15) * 16;             // in halves (32B per thread)
    const bf16_t* gB0 = wB + (size_t)bRowT * EMB_DIM + bColT;  // + block*16*256

    // ---- prime B buffer 0 (async DMA overlaps with the A-tile staging) ----
#if HAVE_ASYNC
    {
        bf16_t* l = &sB[0][bRowT * PITCH + bColT];
        async_cp_b128(gB0,     l);
        async_cp_b128(gB0 + 8, l + 8);
    }
#endif

    // ---- stage A tile [128 rows x 256 ch] as bf16 + fp32 row norms --------
    {
        const int r  = t & 127;           // fixed row per thread
        const int ch = t >> 7;            // 0|1 sub-phase
        float acc = 0.f;
        #pragma unroll 4
        for (int i = 0; i < 128; ++i) {
            int c = 2 * i + ch;
            float x = inB[(size_t)c * SPATIAL + sp0 + r];
            acc += x * x;
            sA[r * PITCH + c] = (__bf16)x;
        }
        sPart[t] = acc;
    }
    __syncthreads();
    if (t < ROWS_PER_WG) sRowN[t] = sPart[t] + sPart[t + 128];

#if HAVE_ASYNC
    __builtin_amdgcn_s_wait_asynccnt(0);
#else
    {
        const uint4* src = (const uint4*)gB0;
        uint4 p0 = src[0], p1 = src[1];
        uint4* dst = (uint4*)(&sB[0][bRowT * PITCH + bColT]);
        dst[0] = p0; dst[1] = p1;
    }
#endif
    __syncthreads();

    // per-lane row norms for this wave's 16-row tile (C-layout rows)
    float rn[8];
    #pragma unroll
    for (int v = 0; v < 8; ++v) rn[v] = sRowN[wave * 16 + half * 8 + v];

    // ---- pin all 8 A fragments (K=256) in VGPRs for the whole loop --------
    // A layout: lanes 0-15 hold K={0-7,16-23}, lanes 16-31 K={8-15,24-31}
    const bf16_t* aRow = &sA[(wave * 16 + m) * PITCH];
    v16bf aF[8];
    #pragma unroll
    for (int kc = 0; kc < 8; ++kc) {
        v8bf a0 = *(const v8bf*)(aRow + kc * 32 + (half ? 8  : 0));
        v8bf a1 = *(const v8bf*)(aRow + kc * 32 + (half ? 24 : 16));
        aF[kc] = __builtin_shufflevector(a0, a1,
            0,1,2,3,4,5,6,7,8,9,10,11,12,13,14,15);
    }

    float minV[8];
    int   minI[8];
    #pragma unroll
    for (int v = 0; v < 8; ++v) { minV[v] = 3.4e38f; minI[v] = 0; }

    // ---- main loop over 64 column blocks of 16 codebook entries -----------
    for (int j = 0; j < NBLK; ++j) {
        const int buf = j & 1;
        const bool pre = (j + 1) < NBLK;

#if HAVE_ASYNC
        if (pre) {   // kick DMA for next block into the back buffer
            const bf16_t* g = gB0 + (size_t)(j + 1) * 16 * EMB_DIM;
            bf16_t* l = &sB[buf ^ 1][bRowT * PITCH + bColT];
            async_cp_b128(g,     l);
            async_cp_b128(g + 8, l + 8);
        }
#else
        uint4 p0, p1;
        if (pre) {
            const uint4* src = (const uint4*)(gB0 + (size_t)(j + 1) * 16 * EMB_DIM);
            p0 = src[0]; p1 = src[1];
        }
#endif
        float cn = cn2[j * 16 + m];                    // column norm (L2-hit)

        v8f acc = {};
        const bf16_t* bRow = &sB[buf][m * PITCH];
        #pragma unroll
        for (int kc = 0; kc < 8; ++kc) {
            // B fragment: lanes 0-15 K={0-7,16-23}, lanes 16-31 K={8-15,24-31}
            v8bf b0 = *(const v8bf*)(bRow + kc * 32 + half * 8);
            v8bf b1 = *(const v8bf*)(bRow + kc * 32 + 16 + half * 8);
            v16bf bb = __builtin_shufflevector(b0, b1,
                0,1,2,3,4,5,6,7,8,9,10,11,12,13,14,15);
            acc = __builtin_amdgcn_wmma_f32_16x16x32_bf16(
                false, aF[kc], false, bb, (short)0, acc, false, false);
        }

        // fuse distance + running argmin
        const int col = j * 16 + m;
        #pragma unroll
        for (int v = 0; v < 8; ++v) {
            float d = rn[v] + cn - 2.0f * acc[v];
            if (d < minV[v]) { minV[v] = d; minI[v] = col; }
        }

#if HAVE_ASYNC
        __builtin_amdgcn_s_wait_asynccnt(0);   // back-buffer DMAs landed
#else
        if (pre) {
            uint4* dst = (uint4*)(&sB[buf ^ 1][bRowT * PITCH + bColT]);
            dst[0] = p0; dst[1] = p1;
        }
#endif
        __syncthreads();
    }

    // ---- cross-lane (16-lane group) min+index reduction -------------------
    #pragma unroll
    for (int v = 0; v < 8; ++v) {
        float bv = minV[v];
        int   bi = minI[v];
        #pragma unroll
        for (int off = 8; off >= 1; off >>= 1) {
            float ov = __shfl_xor(bv, off, 32);
            int   oi = __shfl_xor(bi, off, 32);
            if (ov < bv || (ov == bv && oi < bi)) { bv = ov; bi = oi; }
        }
        if (m == 0) {
            int rowInTile = wave * 16 + half * 8 + v;
            sIdx[rowInTile] = bi;
            outIdx[rBase + rowInTile] = bi;
        }
    }
    __syncthreads();

    // ---- gather quantized vectors from exact fp32 codebook ----------------
    {
        const int r  = t & 127;
        const int ch = t >> 7;
        const int myIdx = sIdx[r];
        const float* wRow = weight + (size_t)myIdx * EMB_DIM;
        float* outB = outQ + (size_t)b * EMB_DIM * SPATIAL + sp0 + r;
        #pragma unroll 4
        for (int i = 0; i < 128; ++i) {
            int c = 2 * i + ch;
            outB[(size_t)c * SPATIAL] = wRow[c];   // contiguous across threads
        }
    }
}

// ---------------------------------------------------------------------------
extern "C" void kernel_launch(void* const* d_in, const int* in_sizes, int n_in,
                              void* d_out, int out_size, void* d_ws, size_t ws_size,
                              hipStream_t stream) {
    const float* inputs = (const float*)d_in[0];   // [32,256,32,32] fp32
    const float* weight = (const float*)d_in[1];   // [1024,256] fp32

    bf16_t* wB  = (bf16_t*)d_ws;                                   // 512 KB
    float*  cn2 = (float*)((char*)d_ws + (size_t)NUM_EMB * EMB_DIM * sizeof(bf16_t));

    float* outQ  = (float*)d_out;                  // 8,388,608 f32
    int*   outIx = (int*)d_out + (size_t)32 * EMB_DIM * SPATIAL;   // 32768 int32

    vq_prep_kernel<<<NUM_EMB, 32, 0, stream>>>(weight, wB, cn2);
    vq_main_kernel<<<N_ROWS / ROWS_PER_WG, 256, 0, stream>>>(
        inputs, weight, wB, cn2, outQ, outIx);
}
